// Contrastive_FeatureExtractor_conv_23794118820188
// MI455X (gfx1250) — compile-verified, matrix operand order checked by reading, not run on hardware
//
#include <hip/hip_runtime.h>

// ---------------------------------------------------------------------------
// Sizes from the reference
#define NROW 4096
#define TLEN 2016
#define L1   2004      // conv1 output length (2016-13+1)
#define L2   167       // conv2 output length (stride 12, K 12)
#define L3   6         // conv3 output length (stride 24, K 24)
#define NH   3
#define EPSV 1e-5f

typedef __attribute__((ext_vector_type(16))) __bf16 v16bf;
typedef __attribute__((ext_vector_type(8)))  float  v8f;
typedef __attribute__((ext_vector_type(4)))  unsigned int u32x4;
typedef __attribute__((ext_vector_type(8)))  int i32x8;
typedef __attribute__((ext_vector_type(4)))  int i32x4;

// WMMA 16x16x32 bf16 -> f32, per probe-confirmed signature
__device__ __forceinline__ v8f wmma_bf16(v16bf a, v16bf b, v8f c) {
  return __builtin_amdgcn_wmma_f32_16x16x32_bf16(false, a, false, b, (short)0, c,
                                                 false, false);
}

// K-index inside a 32-wide K step for 16-bit A/B fragments (ISA 7.12.2)
__device__ __forceinline__ constexpr int kmap16(int e, int half) {
  int i = e >> 1, j = e & 1;
  return ((i >= 4) ? 16 : 0) + 8 * half + 2 * (i & 3) + j;
}

// ---------------------------------------------------------------------------
// Tensor Data Mover: 1-D tile load Global -> LDS.
// Descriptor per CDNA5 ISA. nelem in data_size units
// (dsize_code: 0=1B,1=2B,2=4B,3=8B), nelem <= 65535 so tile_dim0 fits 16 bits.
// 6-arg builtin on this toolchain: (g0, g1, g2, g3, g_extra, cpol);
// groups 2/3/extra unused for tensors up to 2-D -> zeros.
__device__ __forceinline__ void tdm_load_1d(const void* gsrc, void* lds_dst,
                                            unsigned nelem, unsigned dsize_code) {
  unsigned long long ga = (unsigned long long)(uintptr_t)gsrc;
  unsigned lo = (unsigned)(uintptr_t)lds_dst;  // low 32 bits = LDS byte offset
  u32x4 g0;
  g0[0] = 1u;                                       // count=1 (valid user D#)
  g0[1] = lo;                                       // lds_addr
  g0[2] = (unsigned)ga;                             // global_addr[31:0]
  g0[3] = (unsigned)((ga >> 32) & 0x01ffffffu)      // global_addr[56:32]
          | (2u << 30);                             // type=2 ("image")
  i32x8 g1;
  g1[0] = (int)(dsize_code << 16);                  // wg_mask=0, data_size
  g1[1] = (int)((nelem & 0xffffu) << 16);           // tensor_dim0[15:0]
  g1[2] = (int)(((nelem >> 16) & 0xffffu) | (1u << 16)); // dim0 hi | tensor_dim1=1
  g1[3] = (int)((nelem & 0xffffu) << 16);           // tensor_dim1 hi=0 | tile_dim0
  g1[4] = 1;                                        // tile_dim1=1, tile_dim2=0
  g1[5] = (int)nelem;                               // tensor_dim0_stride[31:0]
  g1[6] = 0;                                        // stride0 hi | dim1_stride lo
  g1[7] = 0;
  i32x4 z4 = {0, 0, 0, 0};
  i32x8 z8 = {0, 0, 0, 0, 0, 0, 0, 0};
  __builtin_amdgcn_tensor_load_to_lds(g0, g1, z4, z4, z8, 0);
}

// conv1 single output element (+bias, +relu) from LDS-staged x row
__device__ __forceinline__ float conv1_relu(const float* sx, const float* sw1,
                                            const float* sb1, int c, int t) {
  float acc = sb1[c];
#pragma unroll
  for (int j = 0; j < 13; ++j) acc = fmaf(sw1[c * 13 + j], sx[t + j], acc);
  return fmaxf(acc, 0.f);
}

// ---------------------------------------------------------------------------
// K1: per-row partial (sum, sumsq) of relu(conv1(x)) per channel -> part1[n][64]
__global__ __launch_bounds__(256) void k_stats1(const float* __restrict__ x,
                                                const float* __restrict__ w1,
                                                const float* __restrict__ b1,
                                                float* __restrict__ part1) {
  __shared__ __align__(16) float sx[TLEN];
  __shared__ float sw1[32 * 13];
  __shared__ float sb1[32];
  __shared__ float r1[256], r2[256];
  int n = blockIdx.x, tid = threadIdx.x;
  if (tid < 32) {  // wave 0 drives the TDM
    tdm_load_1d(&x[(size_t)n * TLEN], sx, TLEN, 2);
    __builtin_amdgcn_s_wait_tensorcnt(0);
  }
  for (int t = tid; t < 32 * 13; t += 256) sw1[t] = w1[t];
  if (tid < 32) sb1[tid] = b1[tid];
  __syncthreads();

  int c = tid & 31, grp = tid >> 5;  // 8 position-groups per channel
  float s = 0.f, s2 = 0.f;
  for (int t = grp; t < L1; t += 8) {
    float v = conv1_relu(sx, sw1, sb1, c, t);
    s += v;
    s2 += v * v;
  }
  r1[tid] = s; r2[tid] = s2;
  __syncthreads();
  for (int off = 128; off >= 32; off >>= 1) {
    if (tid < off) { r1[tid] += r1[tid + off]; r2[tid] += r2[tid + off]; }
    __syncthreads();
  }
  if (tid < 32) {
    part1[(size_t)n * 64 + c] = r1[tid];
    part1[(size_t)n * 64 + 32 + c] = r2[tid];
  }
}

// ---------------------------------------------------------------------------
// Column-sum of a [rows x width] matrix (deterministic reduction)
__global__ __launch_bounds__(256) void k_colsum(const float* __restrict__ mat,
                                                float* __restrict__ out, int rows,
                                                int width) {
  int i = blockIdx.x, tid = threadIdx.x;
  float s = 0.f;
  for (int b = tid; b < rows; b += 256) s += mat[(size_t)b * width + i];
  __shared__ float red[256];
  red[tid] = s;
  __syncthreads();
  for (int off = 128; off; off >>= 1) {
    if (tid < off) red[tid] += red[tid + off];
    __syncthreads();
  }
  if (!tid) out[i] = red[0];
}

// Column (sum, sumsq) of a [rows x width] matrix -> out[2i], out[2i+1]
__global__ __launch_bounds__(256) void k_colstats(const float* __restrict__ mat,
                                                  float* __restrict__ out, int rows,
                                                  int width) {
  int i = blockIdx.x, tid = threadIdx.x;
  float s = 0.f, s2 = 0.f;
  for (int b = tid; b < rows; b += 256) {
    float v = mat[(size_t)b * width + i];
    s += v;
    s2 += v * v;
  }
  __shared__ float r1[256], r2[256];
  r1[tid] = s; r2[tid] = s2;
  __syncthreads();
  for (int off = 128; off; off >>= 1) {
    if (tid < off) { r1[tid] += r1[tid + off]; r2[tid] += r2[tid + off]; }
    __syncthreads();
  }
  if (!tid) { out[2 * i] = r1[0]; out[2 * i + 1] = r2[0]; }
}

// ---------------------------------------------------------------------------
// Fold BN (batch stats over `cnt` elems) of a GEMM's input into its weights and
// bias; emit pre-swizzled bf16 WMMA A-fragments.
// afrag layout: [((mt*nsteps + s)*32 + lane)*16 + e]
__global__ __launch_bounds__(256) void k_fold(const float* __restrict__ sums, int sstr,
                                              const float* __restrict__ sqs, int qstr,
                                              float cnt, int Cin, int Kk, int Ktot,
                                              int nsteps, const float* __restrict__ g,
                                              const float* __restrict__ be,
                                              const float* __restrict__ w,
                                              const float* __restrict__ bsrc,
                                              float* __restrict__ bdst,
                                              __bf16* __restrict__ afrag) {
  __shared__ float scale[96], shift[96];
  int tid = threadIdx.x;
  if (tid < Cin) {
    float mu = sums[tid * sstr] / cnt;
    float var = sqs[tid * qstr] / cnt - mu * mu;  // biased var, matches jnp.var
    float inv = rsqrtf(var + EPSV);
    scale[tid] = g[tid] * inv;
    shift[tid] = be[tid] - mu * g[tid] * inv;
  }
  __syncthreads();
  if (tid < 32) {
    float acc = bsrc[tid];
    for (int K = 0; K < Ktot; ++K) acc += w[tid * Ktot + K] * shift[K / Kk];
    bdst[tid] = acc;
  }
  int total = 2 * nsteps * 512;
  for (int idx = tid; idx < total; idx += blockDim.x) {
    int e = idx & 15;
    int lane = (idx >> 4) & 31;
    int rest = idx >> 9;
    int s = rest % nsteps;
    int mt = rest / nsteps;
    int half = lane >> 4;
    int M = mt * 16 + (lane & 15);
    int K = 32 * s + kmap16(e, half);
    afrag[idx] = (__bf16)(w[M * Ktot + K] * scale[K / Kk]);
  }
}

// ---------------------------------------------------------------------------
// conv1 (on the fly, BN1 folded into A) -> conv2 via WMMA -> relu(h2) to LDS.
// saf2: A-fragments staged in LDS. 128 threads = 4 waves; each wave owns
// 16-position N-tiles (11 tiles cover 167 positions). k-loop fully unrolled.
// Out-of-range positions are CLAMPED (pc=0) instead of predicated: B columns
// are independent in WMMA, so garbage in dead columns never reaches a live
// output, and we avoid per-element EXEC save/restore.
__device__ __forceinline__ void conv12_to_lds(const float* sx, const float* sw1,
                                              const float* sb1, const __bf16* saf2,
                                              const float* __restrict__ b2f,
                                              __bf16 (*sh2)[168], int tid) {
  int lane = tid & 31, wave = tid >> 5;
  int half = lane >> 4, col = lane & 15;
  for (int tile = wave; tile < 11; tile += 4) {  // wave-uniform trip count
    int p = tile * 16 + col;                     // conv2 output position
    int t0 = 12 * ((p < L2) ? p : 0);            // clamped base time index
    v8f c0a = {}, c0b = {}, c1a = {}, c1b = {};
#pragma unroll
    for (int s = 0; s < 12; ++s) {               // K = 384 = 12 steps of 32
      v16bf bf;
#pragma unroll
      for (int e = 0; e < 16; ++e) {
        int K = 32 * s + kmap16(e, half);        // constants after unroll
        int ci = K / 12, k = K - ci * 12;
        bf[e] = (__bf16)conv1_relu(sx, sw1, sb1, ci, t0 + k);
      }
      v16bf a0 = *(const v16bf*)(saf2 + ((0 * 12 + s) * 32 + lane) * 16);
      v16bf a1 = *(const v16bf*)(saf2 + ((1 * 12 + s) * 32 + lane) * 16);
      if (s & 1) {
        c0b = wmma_bf16(a0, bf, c0b);
        c1b = wmma_bf16(a1, bf, c1b);
      } else {
        c0a = wmma_bf16(a0, bf, c0a);
        c1a = wmma_bf16(a1, bf, c1a);
      }
    }
    v8f c0 = c0a + c0b, c1 = c1a + c1b;
    if (p < L2) {
#pragma unroll
      for (int r = 0; r < 8; ++r) {
        int m0 = r + 8 * half;                   // C layout: vgpr r -> row r+8*half
        sh2[m0][p]      = (__bf16)fmaxf(c0[r] + b2f[m0], 0.f);
        sh2[m0 + 16][p] = (__bf16)fmaxf(c1[r] + b2f[m0 + 16], 0.f);
      }
    }
  }
}

__device__ __forceinline__ void stage_w1(const float* __restrict__ w1,
                                         const float* __restrict__ b1, float* sw1,
                                         float* sb1, int tid, int nthr) {
  for (int t = tid; t < 32 * 13; t += nthr) sw1[t] = w1[t];
  if (tid < 32) sb1[tid] = b1[tid];
}

// ---------------------------------------------------------------------------
// K2: conv1+conv2 (WMMA), then per-row partial (sum, sumsq) of relu(h2)
__global__ __launch_bounds__(128) void k_conv2_stats(const float* __restrict__ x,
                                                     const float* __restrict__ w1,
                                                     const float* __restrict__ b1,
                                                     const __bf16* __restrict__ afrag2,
                                                     const float* __restrict__ b2f,
                                                     float* __restrict__ part2) {
  __shared__ __align__(16) float sx[TLEN];
  __shared__ float sw1[32 * 13];
  __shared__ float sb1[32];
  __shared__ __align__(16) __bf16 saf2[2 * 12 * 512];
  __shared__ __bf16 sh2[32][168];
  __shared__ float r1[128], r2[128];
  int n = blockIdx.x, tid = threadIdx.x;
  if (tid < 32) {  // wave 0 drives both DMAs
    tdm_load_1d(&x[(size_t)n * TLEN], sx, TLEN, 2);
    tdm_load_1d(afrag2, saf2, 2 * 12 * 512 * 2 / 8, 3);  // 8B units
    __builtin_amdgcn_s_wait_tensorcnt(0);
  }
  stage_w1(w1, b1, sw1, sb1, tid, 128);
  __syncthreads();
  conv12_to_lds(sx, sw1, sb1, saf2, b2f, sh2, tid);
  __syncthreads();

  int c = tid & 31, grp = tid >> 5;  // 4 position-groups
  float s = 0.f, s2 = 0.f;
  for (int p = grp; p < L2; p += 4) {
    float v = (float)sh2[c][p];
    s += v;
    s2 += v * v;
  }
  r1[tid] = s; r2[tid] = s2;
  __syncthreads();
  if (tid < 64) { r1[tid] += r1[tid + 64]; r2[tid] += r2[tid + 64]; }
  __syncthreads();
  if (tid < 32) {
    part2[(size_t)n * 64 + c] = r1[tid] + r1[tid + 32];
    part2[(size_t)n * 64 + 32 + c] = r2[tid] + r2[tid + 32];
  }
}

// ---------------------------------------------------------------------------
// K3: conv1+conv2 (recompute) -> conv3 (WMMA, BN2 folded) -> masked
//     mean/std/max -> feat[n][96]
__global__ __launch_bounds__(128) void k_conv3_feat(
    const float* __restrict__ x, const float* __restrict__ w1,
    const float* __restrict__ b1, const __bf16* __restrict__ afrag2,
    const float* __restrict__ b2f, const __bf16* __restrict__ afrag3,
    const float* __restrict__ b3f, const unsigned char* __restrict__ mask,
    float* __restrict__ feat) {
  __shared__ __align__(16) float sx[TLEN];
  __shared__ float sw1[32 * 13];
  __shared__ float sb1[32];
  __shared__ __align__(16) __bf16 saf2[2 * 12 * 512];
  __shared__ __align__(16) __bf16 saf3[2 * 24 * 512];
  __shared__ __bf16 sh2[32][168];
  __shared__ float sh3[32][L3];
  int n = blockIdx.x, tid = threadIdx.x;
  if (tid < 32) {
    tdm_load_1d(&x[(size_t)n * TLEN], sx, TLEN, 2);
    tdm_load_1d(afrag2, saf2, 2 * 12 * 512 * 2 / 8, 3);
    tdm_load_1d(afrag3, saf3, 2 * 24 * 512 * 2 / 8, 3);
    __builtin_amdgcn_s_wait_tensorcnt(0);
  }
  stage_w1(w1, b1, sw1, sb1, tid, 128);
  __syncthreads();
  conv12_to_lds(sx, sw1, sb1, saf2, b2f, sh2, tid);
  __syncthreads();

  int lane = tid & 31, wave = tid >> 5;
  int half = lane >> 4, col = lane & 15;
  if (wave == 0) {  // single N-tile (6 valid columns), K = 768 = 24 steps
    int zt = 24 * ((col < L3) ? col : 0);  // clamped column -> base position
    v8f c0a = {}, c0b = {}, c1a = {}, c1b = {};
#pragma unroll
    for (int s = 0; s < 24; ++s) {
      v16bf bf;
#pragma unroll
      for (int e = 0; e < 16; ++e) {
        int K = 32 * s + kmap16(e, half);
        int ci = K / 24, k = K - ci * 24;  // constants after unroll
        bf[e] = sh2[ci][zt + k];           // t <= 143 < 167, always in-bounds
      }
      v16bf a0 = *(const v16bf*)(saf3 + ((0 * 24 + s) * 32 + lane) * 16);
      v16bf a1 = *(const v16bf*)(saf3 + ((1 * 24 + s) * 32 + lane) * 16);
      if (s & 1) {
        c0b = wmma_bf16(a0, bf, c0b);
        c1b = wmma_bf16(a1, bf, c1b);
      } else {
        c0a = wmma_bf16(a0, bf, c0a);
        c1a = wmma_bf16(a1, bf, c1a);
      }
    }
    v8f c0 = c0a + c0b, c1 = c1a + c1b;
    if (col < L3) {
#pragma unroll
      for (int r = 0; r < 8; ++r) {
        int m0 = r + 8 * half;
        sh3[m0][col] = c0[r] + b3f[m0];
        sh3[m0 + 16][col] = c1[r] + b3f[m0 + 16];
      }
    }
  }
  __syncthreads();

  if (tid < 32) {
    const unsigned char* mrow = mask + (size_t)n * L3;
    float mu = 0.f;
#pragma unroll
    for (int z = 0; z < L3; ++z)
      if (mrow[z]) mu += sh3[tid][z];
    mu *= (1.f / NH);
    float var = 0.f, mx = -INFINITY;
#pragma unroll
    for (int z = 0; z < L3; ++z)
      if (mrow[z]) {
        float d = sh3[tid][z] - mu;
        var += d * d;
        mx = fmaxf(mx, sh3[tid][z]);
      }
    var *= (1.f / (NH - 1));  // unbiased, matches torch.std semantics
    feat[(size_t)n * 96 + tid] = mu;
    feat[(size_t)n * 96 + 32 + tid] = sqrtf(var);
    feat[(size_t)n * 96 + 64 + tid] = mx;
  }
}

// ---------------------------------------------------------------------------
// K5: fc1 GEMM [4096x96]@[96x32]^T via WMMA (BN3 folded), relu
__global__ __launch_bounds__(32) void k_fc(const float* __restrict__ feat,
                                           const __bf16* __restrict__ afragF,
                                           const float* __restrict__ bFf,
                                           float* __restrict__ fcrelu) {
  int lane = threadIdx.x;
  int half = lane >> 4, col = lane & 15;
  int n = blockIdx.x * 16 + col;  // 4096/16 = 256 blocks, exact
  v8f c0 = {}, c1 = {};
#pragma unroll
  for (int s = 0; s < 3; ++s) {  // K = 96
    v16bf bf;
#pragma unroll
    for (int e = 0; e < 16; ++e) {
      int f = 32 * s + kmap16(e, half);
      bf[e] = (__bf16)feat[(size_t)n * 96 + f];
    }
    v16bf a0 = *(const v16bf*)(afragF + ((size_t)(0 * 3 + s) * 32 + lane) * 16);
    v16bf a1 = *(const v16bf*)(afragF + ((size_t)(1 * 3 + s) * 32 + lane) * 16);
    c0 = wmma_bf16(a0, bf, c0);
    c1 = wmma_bf16(a1, bf, c1);
  }
#pragma unroll
  for (int r = 0; r < 8; ++r) {
    int o = r + 8 * half;
    fcrelu[(size_t)n * 32 + o] = fmaxf(c0[r] + bFf[o], 0.f);
    fcrelu[(size_t)n * 32 + o + 16] = fmaxf(c1[r] + bFf[o + 16], 0.f);
  }
}

// K7: apply final BN over N on relu(fc) using stats4 [2o]=sum, [2o+1]=sumsq
__global__ __launch_bounds__(256) void k_bn4(const float* __restrict__ fcrelu,
                                             const float* __restrict__ stats4,
                                             const float* __restrict__ g4,
                                             const float* __restrict__ be4,
                                             float* __restrict__ out) {
  int idx = blockIdx.x * 256 + threadIdx.x;  // 4096*32 total
  int o = idx & 31;
  float mu = stats4[2 * o] * (1.f / NROW);
  float var = stats4[2 * o + 1] * (1.f / NROW) - mu * mu;
  float inv = rsqrtf(var + EPSV);
  out[idx] = (fcrelu[idx] - mu) * inv * g4[o] + be4[o];
}

// ---------------------------------------------------------------------------
extern "C" void kernel_launch(void* const* d_in, const int* in_sizes, int n_in,
                              void* d_out, int out_size, void* d_ws, size_t ws_size,
                              hipStream_t stream) {
  (void)in_sizes; (void)n_in; (void)out_size; (void)ws_size;
  const float* x = (const float*)d_in[0];
  const unsigned char* mask = (const unsigned char*)d_in[1];  // jnp bool -> 1 byte
  const float* w1 = (const float*)d_in[2];
  const float* b1 = (const float*)d_in[3];
  const float* w2 = (const float*)d_in[4];
  const float* b2 = (const float*)d_in[5];
  const float* w3 = (const float*)d_in[6];
  const float* b3 = (const float*)d_in[7];
  const float* fw = (const float*)d_in[8];
  const float* fb = (const float*)d_in[9];
  const float* g1 = (const float*)d_in[10];
  const float* be1 = (const float*)d_in[11];
  const float* g2 = (const float*)d_in[12];
  const float* be2 = (const float*)d_in[13];
  const float* g3 = (const float*)d_in[14];
  const float* be3 = (const float*)d_in[15];
  const float* g4 = (const float*)d_in[16];
  const float* be4 = (const float*)d_in[17];
  float* out = (float*)d_out;

  char* ws = (char*)d_ws;
  size_t off = 0;
  auto alloc = [&](size_t bytes) -> void* {
    off = (off + 255) & ~(size_t)255;
    void* p = ws + off;
    off += bytes;
    return p;
  };
  float* part1  = (float*)alloc((size_t)NROW * 64 * 4);
  float* stats1 = (float*)alloc(64 * 4);
  float* b2f    = (float*)alloc(32 * 4);
  __bf16* afrag2 = (__bf16*)alloc((size_t)2 * 12 * 512 * 2);
  float* part2  = (float*)alloc((size_t)NROW * 64 * 4);
  float* stats2 = (float*)alloc(64 * 4);
  float* b3f    = (float*)alloc(32 * 4);
  __bf16* afrag3 = (__bf16*)alloc((size_t)2 * 24 * 512 * 2);
  float* feat   = (float*)alloc((size_t)NROW * 96 * 4);
  float* stats3 = (float*)alloc(96 * 2 * 4);
  float* bFf    = (float*)alloc(32 * 4);
  __bf16* afragF = (__bf16*)alloc((size_t)2 * 3 * 512 * 2);
  float* fcrelu = (float*)alloc((size_t)NROW * 32 * 4);
  float* stats4 = (float*)alloc(32 * 2 * 4);

  // Stage 1: BN1 stats of relu(conv1)
  k_stats1<<<NROW, 256, 0, stream>>>(x, w1, b1, part1);
  k_colsum<<<64, 256, 0, stream>>>(part1, stats1, NROW, 64);
  // Fold BN1 into conv2 weights (K=384, 12 steps, ci = K/12)
  k_fold<<<1, 256, 0, stream>>>(stats1, 1, stats1 + 32, 1, (float)NROW * L1, 32, 12,
                                384, 12, g1, be1, w2, b2, b2f, afrag2);
  // Stage 2: BN2 stats of relu(conv2)
  k_conv2_stats<<<NROW, 128, 0, stream>>>(x, w1, b1, afrag2, b2f, part2);
  k_colsum<<<64, 256, 0, stream>>>(part2, stats2, NROW, 64);
  // Fold BN2 into conv3 weights (K=768, 24 steps, ci = K/24)
  k_fold<<<1, 256, 0, stream>>>(stats2, 1, stats2 + 32, 1, (float)NROW * L2, 32, 24,
                                768, 24, g2, be2, w3, b3, b3f, afrag3);
  // Stage 3: conv3 + masked mean/std/max -> feat
  k_conv3_feat<<<NROW, 128, 0, stream>>>(x, w1, b1, afrag2, b2f, afrag3, b3f, mask,
                                         feat);
  // BN3 stats over N, fold into fc weights (K=96, 3 steps, ci = K)
  k_colstats<<<96, 256, 0, stream>>>(feat, stats3, NROW, 96);
  k_fold<<<1, 256, 0, stream>>>(stats3, 2, stats3 + 1, 2, (float)NROW, 96, 1, 96, 3,
                                g3, be3, fw, fb, bFf, afragF);
  // FC GEMM + relu
  k_fc<<<NROW / 16, 32, 0, stream>>>(feat, afragF, bFf, fcrelu);
  // BN4 stats + apply
  k_colstats<<<32, 256, 0, stream>>>(fcrelu, stats4, NROW, 32);
  k_bn4<<<(NROW * 32) / 256, 256, 0, stream>>>(fcrelu, stats4, g4, be4, out);
}
